// UniEGNNConv_8658654068867
// MI455X (gfx1250) — matrix operand order repaired and checked by reading, not compiled, not run on hardware
//
#include <hip/hip_runtime.h>

#define NV_ 100000
#define NE_ 50000
#define DD 128
#define NNZ_ 1600000

typedef __attribute__((ext_vector_type(16))) __bf16 v16bf;
typedef __attribute__((ext_vector_type(8)))  float  v8f;

__device__ inline unsigned short f2bf(float f) {
  unsigned int u = __float_as_uint(f);
  u += 0x7FFFu + ((u >> 16) & 1u);   // round-to-nearest-even
  return (unsigned short)(u >> 16);
}

union AFrag { v16bf v; unsigned short u[16]; uint4 q[2]; };
union BFrag { v16bf v; uint4 q[2]; };

// ---------------------------------------------------------------- weights f32 -> bf16
__global__ void cvt_f32_bf16(const float* __restrict__ src,
                             unsigned short* __restrict__ dst, int n) {
  int i = blockIdx.x * blockDim.x + threadIdx.x;
  int stride = gridDim.x * blockDim.x;
  for (; i < n; i += stride) dst[i] = f2bf(src[i]);
}

// ---------------------------------------------------------------- zero fill (float4)
__global__ void zero_f32(float* __restrict__ p, long n4) {
  long i = (long)blockIdx.x * blockDim.x + threadIdx.x;
  long stride = (long)gridDim.x * blockDim.x;
  float4 z = make_float4(0.f, 0.f, 0.f, 0.f);
  for (; i < n4; i += stride) ((float4*)p)[i] = z;
}

// ---------------------------------------------------------------- one K=128 slab of WMMA MACs
// A layout (16-bit 16x32): lane<16 -> K = ks..ks+7, ks+16..ks+23 of row (lane&15)
//                          lane>=16 -> K = ks+8..15, ks+24..31
template <bool BF, int PITCH>
__device__ inline void mma_part(const void* __restrict__ Ap, int arow, int kbase,
                                const unsigned short* __restrict__ sW,
                                int half, int mlo, v8f* acc) {
#pragma unroll
  for (int ks = 0; ks < 128; ks += 32) {
    const int kk = ks + (half ? 8 : 0);
    AFrag a;
    if constexpr (BF) {
      const unsigned short* ar = (const unsigned short*)Ap + (size_t)arow * DD;
      a.q[0] = *(const uint4*)(ar + kk);
      a.q[1] = *(const uint4*)(ar + kk + 16);
    } else {
      const float* ar = (const float*)Ap + (size_t)arow * DD;
      float4 f0 = *(const float4*)(ar + kk);
      float4 f1 = *(const float4*)(ar + kk + 4);
      float4 f2 = *(const float4*)(ar + kk + 16);
      float4 f3 = *(const float4*)(ar + kk + 20);
      a.u[0]  = f2bf(f0.x); a.u[1]  = f2bf(f0.y); a.u[2]  = f2bf(f0.z); a.u[3]  = f2bf(f0.w);
      a.u[4]  = f2bf(f1.x); a.u[5]  = f2bf(f1.y); a.u[6]  = f2bf(f1.z); a.u[7]  = f2bf(f1.w);
      a.u[8]  = f2bf(f2.x); a.u[9]  = f2bf(f2.y); a.u[10] = f2bf(f2.z); a.u[11] = f2bf(f2.w);
      a.u[12] = f2bf(f3.x); a.u[13] = f2bf(f3.y); a.u[14] = f2bf(f3.z); a.u[15] = f2bf(f3.w);
    }
    const int kb = kbase + ks + (half ? 8 : 0);
#pragma unroll
    for (int t = 0; t < 8; ++t) {
      // B fragment 32x16: col n, K=kb..kb+7, kb+16..kb+23 (kb includes +8 for lane>=16)
      const int n = t * 16 + mlo;
      BFrag b;
      const uint4* bp = (const uint4*)(sW + n * PITCH + kb);
      b.q[0] = bp[0];
      b.q[1] = bp[2];
      acc[t] = __builtin_amdgcn_wmma_f32_16x16x32_bf16(
          false, a.v, false, b.v, (short)0, acc[t], false, false);
    }
  }
}

// ---------------------------------------------------------------- WMMA GEMM
// C[M,128] = concat_k(A0,A1)[M, KPARTS*128] x Wb^T + bias
// Wb is bf16, row-major [128, KPARTS*128]. Block: 256 threads = 8 waves,
// each wave owns a 16-row strip x N=128 (8 accumulator tiles).
template <int KPARTS, bool A0BF, bool A1BF, bool OUTBF>
__global__ void __launch_bounds__(256)
gemm_bias(const void* __restrict__ A0, const void* __restrict__ A1,
          const unsigned short* __restrict__ Wb, const float* __restrict__ bias,
          void* __restrict__ Cout, int M) {
  constexpr int K = KPARTS * 128;
  constexpr int PITCH = K + 8;               // 16B row pad: dodge LDS bank conflicts
  extern __shared__ unsigned short sW[];     // [128 * PITCH]
  const int tid = threadIdx.x;

  // cooperative stage of the shared weight tile into LDS
  {
    const int chunksPerRow = K / 8;          // uint4 = 8 bf16
    const int total = 128 * chunksPerRow;
    for (int i = tid; i < total; i += 256) {
      int row = i / chunksPerRow, c = i - row * chunksPerRow;
      ((uint4*)(sW + row * PITCH))[c] = ((const uint4*)(Wb + row * K))[c];
    }
  }
  __syncthreads();

  const int wave = tid >> 5;
  const int lane = tid & 31;
  const int half = lane >> 4;                // wave32: lanes 0-15 vs 16-31
  const int mlo  = lane & 15;
  const int rowBase = blockIdx.x * 128 + wave * 16;
  int arow = rowBase + mlo;
  if (arow >= M) arow = M - 1;               // clamp loads; stores are guarded

  v8f acc[8];
#pragma unroll
  for (int t = 0; t < 8; ++t) acc[t] = v8f{};

  mma_part<A0BF, PITCH>(A0, arow, 0, sW, half, mlo, acc);
  if constexpr (KPARTS == 2) {
    mma_part<A1BF, PITCH>(A1, arow, 128, sW, half, mlo, acc);
  }

  // epilogue: C/D 16x16 layout -> lane<16: M=j, lane>=16: M=8+j ; N = 16t + mlo
  const int mstore = rowBase + (half ? 8 : 0);
#pragma unroll
  for (int t = 0; t < 8; ++t) {
    const int n = t * 16 + mlo;
    const float bn = bias[n];
#pragma unroll
    for (int j = 0; j < 8; ++j) {
      const int m = mstore + j;
      if (m < M) {
        const float r = acc[t][j] + bn;
        if constexpr (OUTBF)
          ((unsigned short*)Cout)[(size_t)m * DD + n] = f2bf(r);
        else
          ((float*)Cout)[(size_t)m * DD + n] = r;
      }
    }
  }
}

// ---------------------------------------------------------------- gather + atomic scatter-add
// One wave per incidence: 32 lanes x float4 = 128 floats, fully coalesced.
__global__ void __launch_bounds__(256)
scatter_accum(const float* __restrict__ src, const int* __restrict__ gidx,
              const int* __restrict__ sidx, float* __restrict__ sum,
              float* __restrict__ cnt, int nnz) {
  const int lane = threadIdx.x & 31;
  const int wave = (blockIdx.x * blockDim.x + threadIdx.x) >> 5;
  const int nWaves = (gridDim.x * blockDim.x) >> 5;
  for (int i = wave; i < nnz; i += nWaves) {
    const int g = gidx[i];
    const int s = sidx[i];
    const int inext = i + nWaves;
    if (inext < nnz) {  // software prefetch of next gathered row (global_prefetch_b8)
      __builtin_prefetch(src + (size_t)gidx[inext] * DD + lane * 4, 0, 1);
    }
    if (lane == 0) unsafeAtomicAdd(cnt + s, 1.0f);
    const float4 v = ((const float4*)(src + (size_t)g * DD))[lane];
    float* dst = sum + (size_t)s * DD + lane * 4;
    unsafeAtomicAdd(dst + 0, v.x);
    unsafeAtomicAdd(dst + 1, v.y);
    unsafeAtomicAdd(dst + 2, v.z);
    unsafeAtomicAdd(dst + 3, v.w);
  }
}

// ---------------------------------------------------------------- sum -> mean, emit bf16 msgs
// (only consumer of the messages is a bf16 WMMA GEMM -> emit bf16 directly)
__global__ void seg_mean_div_bf(const float* __restrict__ sum,
                                const float* __restrict__ cnt,
                                unsigned short* __restrict__ obf, int rows) {
  int i = blockIdx.x * blockDim.x + threadIdx.x;
  int stride = gridDim.x * blockDim.x;
  const int n4 = rows * 32;
  for (; i < n4; i += stride) {
    const int row = i >> 5;
    const float inv = 1.0f / fmaxf(cnt[row], 1.0f);
    float4 v = ((const float4*)sum)[i];
    uint2 o;
    o.x = (unsigned)f2bf(v.x * inv) | ((unsigned)f2bf(v.y * inv) << 16);
    o.y = (unsigned)f2bf(v.z * inv) | ((unsigned)f2bf(v.w * inv) << 16);
    ((uint2*)obf)[i] = o;
  }
}

// ---------------------------------------------------------------- LayerNorm + ReLU (wave per row)
__global__ void __launch_bounds__(256)
ln_relu(const float* __restrict__ Z, const float* __restrict__ g,
        const float* __restrict__ b, float* __restrict__ O, int M) {
  const int lane = threadIdx.x & 31;
  const int wave = (blockIdx.x * blockDim.x + threadIdx.x) >> 5;
  const int nWaves = (gridDim.x * blockDim.x) >> 5;
  const float4 gv = ((const float4*)g)[lane];
  const float4 bv = ((const float4*)b)[lane];
  for (int r = wave; r < M; r += nWaves) {
    float4 v = ((const float4*)(Z + (size_t)r * DD))[lane];
    float s = v.x + v.y + v.z + v.w;
#pragma unroll
    for (int m = 16; m > 0; m >>= 1) s += __shfl_xor(s, m, 32);
    const float mean = s * (1.0f / 128.0f);
    const float dx = v.x - mean, dy = v.y - mean, dz = v.z - mean, dw = v.w - mean;
    float q = dx * dx + dy * dy + dz * dz + dw * dw;
#pragma unroll
    for (int m = 16; m > 0; m >>= 1) q += __shfl_xor(q, m, 32);
    const float rs = rsqrtf(q * (1.0f / 128.0f) + 1e-5f);
    float4 o;
    o.x = fmaxf(dx * rs * gv.x + bv.x, 0.f);
    o.y = fmaxf(dy * rs * gv.y + bv.y, 0.f);
    o.z = fmaxf(dz * rs * gv.z + bv.z, 0.f);
    o.w = fmaxf(dw * rs * gv.w + bv.w, 0.f);
    ((float4*)(O + (size_t)r * DD))[lane] = o;
  }
}

// ================================================================ host
extern "C" void kernel_launch(void* const* d_in, const int* in_sizes, int n_in,
                              void* d_out, int out_size, void* d_ws, size_t ws_size,
                              hipStream_t stream) {
  (void)in_sizes; (void)n_in; (void)out_size; (void)ws_size;
  const float* X    = (const float*)d_in[0];
  const float* Y    = (const float*)d_in[1];
  const int*  v_idx = (const int*)d_in[2];
  const int*  e_idx = (const int*)d_in[3];
  const float* W_tv = (const float*)d_in[4];
  const float* b_tv = (const float*)d_in[5];
  const float* W_te = (const float*)d_in[6];
  const float* b_te = (const float*)d_in[7];
  const float* W_em = (const float*)d_in[8];
  const float* b_em = (const float*)d_in[9];
  const float* W_vm = (const float*)d_in[10];
  const float* b_vm = (const float*)d_in[11];
  const float* g_v  = (const float*)d_in[12];
  const float* be_v = (const float*)d_in[13];
  const float* g_e  = (const float*)d_in[14];
  const float* be_e = (const float*)d_in[15];
  float* out = (float*)d_out;

  // ---- workspace layout (aliased; ~155 MB) ----
  unsigned short* w_tv_bf = (unsigned short*)d_ws;          // 16384
  unsigned short* w_te_bf = w_tv_bf + 16384;                // 16384
  unsigned short* w_em_bf = w_tv_bf + 32768;                // 32768
  unsigned short* w_vm_bf = w_tv_bf + 65536;                // 32768 (ends 196608 B)
  unsigned short* msgE_bf  = (unsigned short*)((char*)d_ws + 262144);   // NE*128 bf16
  unsigned short* msgV_bf  = msgE_bf + (size_t)NE_ * DD;                // NV*128 bf16
  unsigned short* ymMid_bf = msgV_bf + (size_t)NV_ * DD;                // NE*128 bf16
  float* f     = (float*)(ymMid_bf + (size_t)NE_ * DD);
  float* fXp   = f;                              // NV*128 f32 : Xp, later sum_v
  float* fSumE = fXp + (size_t)NV_ * DD;         // NE*128 f32
  float* fCntE = fSumE + (size_t)NE_ * DD;       // NE
  float* fYm   = fCntE + NE_;                    // NE*128 f32 (kept for Yo)
  float* fCntV = fYm + (size_t)NE_ * DD;         // NV
  float* fSumV = fXp;                            // alias: Xp dead after v->e scatter

  const int smem1 = 128 * (128 + 8) * 2;  // 34.0 KB
  const int smem2 = 128 * (256 + 8) * 2;  // 66.0 KB (WGP has 320 KB LDS)

  // 1) weights -> bf16
  cvt_f32_bf16<<<64, 256, 0, stream>>>(W_tv, w_tv_bf, 16384);
  cvt_f32_bf16<<<64, 256, 0, stream>>>(W_te, w_te_bf, 16384);
  cvt_f32_bf16<<<128, 256, 0, stream>>>(W_em, w_em_bf, 32768);
  cvt_f32_bf16<<<128, 256, 0, stream>>>(W_vm, w_vm_bf, 32768);

  // 2) Xp = X @ W_tv^T + b_tv   (f32 out: feeds f32 segment sums)
  gemm_bias<1, false, false, false>
      <<<(NV_ + 127) / 128, 256, smem1, stream>>>(X, nullptr, w_tv_bf, b_tv, fXp, NV_);

  // 3) v -> e mean (bf16 message out)
  zero_f32<<<2048, 256, 0, stream>>>(fSumE, (long)NE_ * DD / 4);
  zero_f32<<<64, 256, 0, stream>>>(fCntE, NE_ / 4);
  scatter_accum<<<4096, 256, 0, stream>>>(fXp, v_idx, e_idx, fSumE, fCntE, NNZ_);
  seg_mean_div_bf<<<2048, 256, 0, stream>>>(fSumE, fCntE, msgE_bf, NE_);

  // 4) Ym_mid = [Y | msg_e] @ W_em^T + b_em  (bf16 out: only consumer is next GEMM)
  gemm_bias<2, false, true, true>
      <<<(NE_ + 127) / 128, 256, smem2, stream>>>(Y, msgE_bf, w_em_bf, b_em, ymMid_bf, NE_);

  // 5) Ym = Ym_mid @ W_te^T + b_te  (f32 out: feeds scatter + LN)
  gemm_bias<1, true, false, false>
      <<<(NE_ + 127) / 128, 256, smem1, stream>>>(ymMid_bf, nullptr, w_te_bf, b_te, fYm, NE_);

  // 6) e -> v mean (sum_v reuses Xp region; bf16 message out)
  zero_f32<<<4096, 256, 0, stream>>>(fSumV, (long)NV_ * DD / 4);
  zero_f32<<<128, 256, 0, stream>>>(fCntV, NV_ / 4);
  scatter_accum<<<4096, 256, 0, stream>>>(fYm, e_idx, v_idx, fSumV, fCntV, NNZ_);
  seg_mean_div_bf<<<4096, 256, 0, stream>>>(fSumV, fCntV, msgV_bf, NV_);

  // 7) Xm = [X | msg_v] @ W_vm^T + b_vm  -> directly into d_out X region
  gemm_bias<2, false, true, false>
      <<<(NV_ + 127) / 128, 256, smem2, stream>>>(X, msgV_bf, w_vm_bf, b_vm, out, NV_);

  // 8) Xo = relu(LN(Xm)) in place ; Yo = relu(LN(Ym))
  ln_relu<<<4096, 256, 0, stream>>>(out, g_v, be_v, out, NV_);
  ln_relu<<<2048, 256, 0, stream>>>(fYm, g_e, be_e, out + (size_t)NV_ * DD, NE_);
}
